// FastCompatibleMultiheadAttention_40716289966621
// MI455X (gfx1250) — matrix operand order repaired and checked by reading, not compiled, not run on hardware
//
#include <hip/hip_runtime.h>

typedef __attribute__((ext_vector_type(16))) __bf16 v16bf;
typedef __attribute__((ext_vector_type(8)))  float  v8f;
typedef __attribute__((ext_vector_type(8)))  int    v8i;

#define WMMA_BF16(a, b, c) \
  __builtin_amdgcn_wmma_f32_16x16x32_bf16(false, (a), false, (b), (short)0, (c), false, false)

__device__ inline unsigned short f2bf(float x) {
  __bf16 h = (__bf16)x;
  return __builtin_bit_cast(unsigned short, h);
}
__device__ inline unsigned int pack2bf(float lo, float hi) {
  return (unsigned)f2bf(lo) | ((unsigned)f2bf(hi) << 16);
}
__device__ inline v8f v8fz() {
  v8f z = {0.f, 0.f, 0.f, 0.f, 0.f, 0.f, 0.f, 0.f};
  return z;
}

// A fragment (16x32 bf16, MxK): lane l holds row m = mBase + l%16.
// K pairs: k0 = 2j + (lane>=16 ? 16:0).
// src row-major [m][k], row stride strideE (elements, even), dword-aligned.
__device__ inline v16bf load_a(const unsigned short* __restrict__ src,
                               size_t strideE, int mBase, int kBase) {
  const int lane = threadIdx.x & 31;
  const int m = mBase + (lane & 15);
  const int hi = lane >> 4;
  const unsigned int* p = (const unsigned int*)(src + (size_t)m * strideE + kBase);
  v8i a;
#pragma unroll
  for (int j = 0; j < 8; ++j) {
    const int k0 = 2 * j + (hi ? 16 : 0);
    a[j] = p[k0 >> 1];
  }
  return __builtin_bit_cast(v16bf, a);
}

// B fragment (32x16 bf16, KxN): lane l holds col n = nBase + l%16.
// K pairs: k0 = 2j + (lane>=16 ? 16:0).
// src stored transposed [n][k], row stride strideE (elements, even).
__device__ inline v16bf load_b(const unsigned short* __restrict__ src,
                               size_t strideE, int nBase, int kBase) {
  const int lane = threadIdx.x & 31;
  const int n = nBase + (lane & 15);
  const int hi = lane >> 4;
  const unsigned int* p = (const unsigned int*)(src + (size_t)n * strideE + kBase);
  v8i a;
#pragma unroll
  for (int j = 0; j < 8; ++j) {
    const int k0 = 2 * j + (hi ? 16 : 0);
    a[j] = p[k0 >> 1];
  }
  return __builtin_bit_cast(v16bf, a);
}

// ---------------------------------------------------------------------------
// fp32 -> bf16 bulk convert (n divisible by 4)
// ---------------------------------------------------------------------------
__global__ __launch_bounds__(256) void cvt_kernel(const float* __restrict__ in,
                                                  unsigned short* __restrict__ out,
                                                  int n) {
  const int i = (blockIdx.x * blockDim.x + threadIdx.x) << 2;
  if (i + 3 < n) {
    const float4 v = *(const float4*)(in + i);
    uint2 o;
    o.x = pack2bf(v.x, v.y);
    o.y = pack2bf(v.z, v.w);
    *(uint2*)(out + i) = o;
  }
}

// ---------------------------------------------------------------------------
// GEMM: C[m][n] = sum_k A[m][k] * Bw[n][k]  (A: MxK bf16, Bw: NxK bf16)
// Block tile 128(M) x 128(N), K-step 32. 8 waves: 4(M) x 2(N), wave tile 32x64.
// MODE 0: qkv epilogue (+bias, q-scale, scatter to Q/K/Vt bf16)
// MODE 1: out-proj epilogue (+bias, fp32 row-major store)
// ---------------------------------------------------------------------------
template <int MODE>
__global__ __launch_bounds__(256) void gemm_kernel(
    const unsigned short* __restrict__ A,
    const unsigned short* __restrict__ Bw,
    const float* __restrict__ bias,
    float* __restrict__ outF,
    unsigned short* __restrict__ Qw,
    unsigned short* __restrict__ Kw,
    unsigned short* __restrict__ Vt,
    int M, int N, int K, int T, int Bsz, int H) {
  __shared__ unsigned short As[128 * 34];
  __shared__ unsigned short Bs[128 * 34];

  const int tid = threadIdx.x;
  const int blockM = blockIdx.y * 128;
  const int blockN = blockIdx.x * 128;
  const int w = tid >> 5;
  const int wm = w & 3;   // M wave index 0..3 (32 rows each)
  const int wn = w >> 2;  // N wave index 0..1 (64 cols each)
  const int lane = tid & 31;
  const int hi = lane >> 4, col = lane & 15;

  v8f acc[2][4];
#pragma unroll
  for (int mi = 0; mi < 2; ++mi)
#pragma unroll
    for (int c = 0; c < 4; ++c) acc[mi][c] = v8fz();

  const int ar = tid >> 2;         // staging row 0..63
  const int ac = (tid & 3) << 3;   // staging col (bf16) 0,8,16,24

  for (int kk = 0; kk < K; kk += 32) {
    __syncthreads();
    // Stage A tile 128x32 and B tile 128x32 (2 x 16B per thread each)
#pragma unroll
    for (int i = 0; i < 2; ++i) {
      const int r = ar + (i << 6);
      const uint4 av = *(const uint4*)(A + (size_t)(blockM + r) * K + kk + ac);
      unsigned int* da = (unsigned int*)&As[r * 34 + ac];
      da[0] = av.x; da[1] = av.y; da[2] = av.z; da[3] = av.w;
      const uint4 bv = *(const uint4*)(Bw + (size_t)(blockN + r) * K + kk + ac);
      unsigned int* db = (unsigned int*)&Bs[r * 34 + ac];
      db[0] = bv.x; db[1] = bv.y; db[2] = bv.z; db[3] = bv.w;
    }
    __syncthreads();

    v16bf af[2];
#pragma unroll
    for (int mi = 0; mi < 2; ++mi) af[mi] = load_a(As, 34, wm * 32 + mi * 16, 0);
#pragma unroll
    for (int c = 0; c < 4; ++c) {
      const v16bf bfr = load_b(Bs, 34, wn * 64 + c * 16, 0);
#pragma unroll
      for (int mi = 0; mi < 2; ++mi)
        acc[mi][c] = WMMA_BF16(af[mi], bfr, acc[mi][c]);
    }
  }

#pragma unroll
  for (int mi = 0; mi < 2; ++mi) {
#pragma unroll
    for (int c = 0; c < 4; ++c) {
#pragma unroll
      for (int r = 0; r < 8; ++r) {
        const int m = blockM + wm * 32 + mi * 16 + r + 8 * hi;
        const int n = blockN + wn * 64 + c * 16 + col;
        float v = acc[mi][c][r] + bias[n];
        if (MODE == 0) {
          // n = h*192 + j3*64 + dd ; m = t*Bsz + b
          const int h = n / 192;
          const int j3 = (n >> 6) % 3;
          const int dd = n & 63;
          const int t = m / Bsz;
          const int bb = m % Bsz;
          const int bh = bb * H + h;
          const size_t idx = ((size_t)bh * T + t) * 64 + dd;
          if (j3 == 0)      Qw[idx] = f2bf(v * 0.125f);  // scaling = d^-0.5
          else if (j3 == 1) Kw[idx] = f2bf(v);
          else              Vt[((size_t)bh * 64 + dd) * T + t] = f2bf(v);
        } else {
          outF[(size_t)m * N + n] = v;
        }
      }
    }
  }
}

// ---------------------------------------------------------------------------
// Flash attention, transposed formulation: one wave per 16-row q-block.
//   S^T(keys x q) = K_tile (A) x Q^T (B);  ctx^T(d x q) = V^T (A) x P^T (B)
// Q/K: [bh][T][64] bf16 (q pre-scaled). V: [bh][64][T] bf16 (transposed).
// q index lives on lanes -> softmax stats are per-lane scalars; P^T B-fragment
// assembled with 8 cross-half shuffles; no LDS at all.
// ---------------------------------------------------------------------------
__global__ __launch_bounds__(256) void flash_kernel(
    const unsigned short* __restrict__ Qw,
    const unsigned short* __restrict__ Kw,
    const unsigned short* __restrict__ Vt,
    unsigned short* __restrict__ ctxW,
    int T, int Bsz, int H) {
  const int tid = threadIdx.x;
  const int w = tid >> 5, lane = tid & 31;
  const int hi = lane >> 4, col = lane & 15;

  const int nQ = T >> 4;  // q-blocks per (b,h)
  const int gw = blockIdx.x * 8 + w;
  const int bh = gw / nQ;
  const int qb = gw % nQ;

  const unsigned short* Qp = Qw + ((size_t)bh * T + (size_t)qb * 16) * 64;
  const unsigned short* Kp = Kw + (size_t)bh * T * 64;
  const unsigned short* Vp = Vt + (size_t)bh * 64 * T;

  // Q^T as B fragments (lane = q row, K pairs = d, contiguous in Qw rows)
  const v16bf bQ0 = load_b(Qp, 64, 0, 0);    // d 0..31
  const v16bf bQ1 = load_b(Qp, 64, 0, 32);   // d 32..63

  v8f ctx[4];  // ctx^T: 4 tiles of (16 d-rows x 16 q-cols)
#pragma unroll
  for (int c = 0; c < 4; ++c) ctx[c] = v8fz();
  float mq = -3.0e38f;  // per-lane running max for q col
  float lq = 0.f;       // per-lane running sum

  for (int s = 0; s < T; s += 32) {
    // S^T tiles: St0 = keys s..s+15, St1 = keys s+16..s+31 (16x16 each)
    v8f St0 = v8fz(), St1 = v8fz();
    St0 = WMMA_BF16(load_a(Kp, 64, s, 0),       bQ0, St0);
    St0 = WMMA_BF16(load_a(Kp, 64, s, 32),      bQ1, St0);
    St1 = WMMA_BF16(load_a(Kp, 64, s + 16, 0),  bQ0, St1);
    St1 = WMMA_BF16(load_a(Kp, 64, s + 16, 32), bQ1, St1);
    // lane holds q col (lane&15); keys on VGPRs/halves:
    //   lo half: St0 -> keys s+r, St1 -> keys s+16+r
    //   hi half: St0 -> keys s+8+r, St1 -> keys s+24+r   (disjoint halves)

    // per-lane max over the 16 local keys, then combine across halves
    float mx = St0[0];
#pragma unroll
    for (int r = 1; r < 8; ++r) mx = fmaxf(mx, St0[r]);
#pragma unroll
    for (int r = 0; r < 8; ++r) mx = fmaxf(mx, St1[r]);
    mx = fmaxf(mx, __shfl_xor(mx, 16, 32));
    const float mnew = fmaxf(mq, mx);
    const float corr = __expf(mq - mnew);
    mq = mnew;

    float e0[8], e1[8];
    float rs = 0.f;
#pragma unroll
    for (int r = 0; r < 8; ++r) {
      e0[r] = __expf(St0[r] - mnew);
      e1[r] = __expf(St1[r] - mnew);
      rs += e0[r] + e1[r];
    }
    rs += __shfl_xor(rs, 16, 32);
    lq = lq * corr + rs;

    // assemble P^T B-fragment: lane needs keys {0..15} (lo) / {16..31} (hi)
    // of its q col; 8 of the 16 live in the partner lane (xor 16).
    float rcv[8];
#pragma unroll
    for (int r = 0; r < 8; ++r) {
      const float send = hi ? e0[r] : e1[r];
      rcv[r] = __shfl_xor(send, 16, 32);
    }
    v8i pk;
#pragma unroll
    for (int j = 0; j < 4; ++j)
      pk[j] = hi ? pack2bf(rcv[2 * j], rcv[2 * j + 1])
                 : pack2bf(e0[2 * j], e0[2 * j + 1]);
#pragma unroll
    for (int j = 4; j < 8; ++j)
      pk[j] = hi ? pack2bf(e1[2 * j - 8], e1[2 * j - 7])
                 : pack2bf(rcv[2 * j - 8], rcv[2 * j - 7]);
    const v16bf bP = __builtin_bit_cast(v16bf, pk);

    // rescale ctx by per-lane corr, then ctx^T += V^T x P^T
#pragma unroll
    for (int c = 0; c < 4; ++c) {
#pragma unroll
      for (int r = 0; r < 8; ++r) ctx[c][r] *= corr;
      ctx[c] = WMMA_BF16(load_a(Vp, (size_t)T, c * 16, s), bP, ctx[c]);
    }
  }

  // finalize + write ctx^T -> ctxW [t][b][e] (8 contiguous bf16 per lane/tile)
  const int h = bh % H, bb = bh / H;
  const int E = H * 64;
  const int t = qb * 16 + col;  // q col on lanes
  const float inv_l = 1.0f / lq;
#pragma unroll
  for (int c = 0; c < 4; ++c) {
    uint4 o;
    o.x = pack2bf(ctx[c][0] * inv_l, ctx[c][1] * inv_l);
    o.y = pack2bf(ctx[c][2] * inv_l, ctx[c][3] * inv_l);
    o.z = pack2bf(ctx[c][4] * inv_l, ctx[c][5] * inv_l);
    o.w = pack2bf(ctx[c][6] * inv_l, ctx[c][7] * inv_l);
    const size_t e = (size_t)h * 64 + c * 16 + 8 * hi;
    *(uint4*)(ctxW + ((size_t)t * Bsz + bb) * E + e) = o;
  }
}

// ---------------------------------------------------------------------------
extern "C" void kernel_launch(void* const* d_in, const int* in_sizes, int n_in,
                              void* d_out, int out_size, void* d_ws, size_t ws_size,
                              hipStream_t stream) {
  constexpr int T = 2048, B = 2, E = 1024, H = 16;
  constexpr int M = T * B;    // 4096
  constexpr int N1 = 3 * E;   // 3072

  const float* query = (const float*)d_in[0];
  const float* winW  = (const float*)d_in[1];
  const float* inB   = (const float*)d_in[2];
  const float* woutW = (const float*)d_in[3];
  const float* outB  = (const float*)d_in[4];

  // workspace layout (bf16), ~48 MB total
  unsigned short* qA     = (unsigned short*)d_ws;
  unsigned short* winBf  = qA     + (size_t)M * E;
  unsigned short* woutBf = winBf  + (size_t)N1 * E;
  unsigned short* Qw     = woutBf + (size_t)E * E;
  unsigned short* Kw     = Qw + (size_t)B * H * T * 64;
  unsigned short* Vt     = Kw + (size_t)B * H * T * 64;
  unsigned short* ctxW   = Vt + (size_t)B * H * T * 64;

  cvt_kernel<<<(M * E / 4 + 255) / 256, 256, 0, stream>>>(query, qA, M * E);
  cvt_kernel<<<(N1 * E / 4 + 255) / 256, 256, 0, stream>>>(winW, winBf, N1 * E);
  cvt_kernel<<<(E * E / 4 + 255) / 256, 256, 0, stream>>>(woutW, woutBf, E * E);

  gemm_kernel<0><<<dim3(N1 / 128, M / 128), 256, 0, stream>>>(
      qA, winBf, inB, nullptr, Qw, Kw, Vt, M, N1, E, T, B, H);

  flash_kernel<<<(B * H * (T / 16)) / 8, 256, 0, stream>>>(Qw, Kw, Vt, ctxW, T, B, H);

  gemm_kernel<1><<<dim3(E / 128, M / 128), 256, 0, stream>>>(
      ctxW, woutBf, outB, (float*)d_out, nullptr, nullptr, nullptr, M, E, E, T, B, H);
}
